// TransformerBlock_40750649704763
// MI455X (gfx1250) — matrix operand order repaired and checked by reading, not compiled, not run on hardware
//
#include <hip/hip_runtime.h>
#include <hip/hip_bf16.h>
#include <math.h>

// ---------------------------------------------------------------------------
// Transformer block for MI455X (gfx1250): f16 WMMA GEMMs (async-LDS staged
// B tiles) + causal flash attention.
// ---------------------------------------------------------------------------

typedef _Float16 f16;
typedef _Float16 v8h  __attribute__((ext_vector_type(8)));
typedef _Float16 v16h __attribute__((ext_vector_type(16)));
typedef float    v8f  __attribute__((ext_vector_type(8)));
typedef int      v4i  __attribute__((ext_vector_type(4)));

static constexpr int Bc  = 2;
static constexpr int Tc  = 2048;
static constexpr int Dc  = 1024;
static constexpr int Hc  = 16;
static constexpr int HWc = 64;
static constexpr int FFc = 4096;
static constexpr int Rc  = Bc * Tc;   // 4096 token rows

#if defined(__HIP_DEVICE_COMPILE__) && \
    __has_builtin(__builtin_amdgcn_global_load_async_to_lds_b128)
#define ASYNC_LDS 1
#else
#define ASYNC_LDS 0
#endif

#if ASYNC_LDS
typedef __attribute__((address_space(1))) v4i* gv4i_p;  // global 16B chunk
typedef __attribute__((address_space(3))) v4i* lv4i_p;  // LDS 16B chunk
#endif

// ---- WMMA wrapper ----------------------------------------------------------
__device__ inline v8f wmma_f16(v16h a, v16h b, v8f c) {
  // D = A(16x32 f16) * B(32x16 f16) + C(16x16 f32)
  return __builtin_amdgcn_wmma_f32_16x16x32_f16(
      /*neg_a=*/false, a, /*neg_b=*/false, b,
      /*c_mod=*/(short)0, c, /*reuse_a=*/false, /*reuse_b=*/false);
}

// Fragment loader from two 16B chunks (works for global and LDS pointers).
__device__ inline v16h ldA(const f16* p_lo, const f16* p_hi) {
  v8h a = *(const v8h*)p_lo;
  v8h b = *(const v8h*)p_hi;
  v16h r;
#pragma unroll
  for (int j = 0; j < 8; ++j) { r[j] = a[j]; r[j + 8] = b[j]; }
  return r;
}

// 16B global -> LDS stage: async copy on gfx1250 (tracked by ASYNCcnt),
// plain load/store fallback elsewhere.
__device__ inline void stage16B(const f16* gsrc, f16* ldst) {
#if ASYNC_LDS
  gv4i_p g = (gv4i_p)(unsigned long long)(uintptr_t)gsrc;
  lv4i_p l = (lv4i_p)(unsigned)(uintptr_t)ldst;
  __builtin_amdgcn_global_load_async_to_lds_b128(g, l, 0, 0);
#else
  *(v8h*)ldst = *(const v8h*)gsrc;
#endif
}

__device__ inline void wait_stage() {
#if ASYNC_LDS
  asm volatile("s_wait_asynccnt 0" ::: "memory");
#endif
}

__device__ inline float gelu_t(float x) {
  const float c = 0.7978845608028654f;  // sqrt(2/pi)
  return 0.5f * x * (1.0f + tanhf(c * (x + 0.044715f * x * x * x)));
}

// ---------------------------------------------------------------------------
// Weight convert+transpose: f32 [K,N] row-major -> f16 [N,K] row-major.
// ---------------------------------------------------------------------------
__global__ void wtrans_f16(const float* __restrict__ in, f16* __restrict__ out,
                           int K, int N) {
  size_t i = (size_t)blockIdx.x * blockDim.x + threadIdx.x;
  if (i >= (size_t)K * N) return;
  int n  = (int)(i / K);
  int kk = (int)(i % K);
  out[i] = (f16)in[(size_t)kk * N + n];
}

// ---------------------------------------------------------------------------
// LayerNorm over D=1024, emit f16 activations. One block per row.
// ---------------------------------------------------------------------------
__global__ __launch_bounds__(256)
void ln_to_f16(const float* __restrict__ x, const float* __restrict__ g,
               const float* __restrict__ s, f16* __restrict__ out) {
  __shared__ float red[256];
  const int row = blockIdx.x, tid = threadIdx.x;
  const float* xr = x + (size_t)row * Dc;
  float v[4];
  float acc = 0.f;
#pragma unroll
  for (int j = 0; j < 4; ++j) { v[j] = xr[tid + j * 256]; acc += v[j]; }
  red[tid] = acc; __syncthreads();
  for (int off = 128; off > 0; off >>= 1) {
    if (tid < off) red[tid] += red[tid + off];
    __syncthreads();
  }
  const float mean = red[0] * (1.0f / Dc);
  __syncthreads();
  acc = 0.f;
#pragma unroll
  for (int j = 0; j < 4; ++j) { float d = v[j] - mean; acc += d * d; }
  red[tid] = acc; __syncthreads();
  for (int off = 128; off > 0; off >>= 1) {
    if (tid < off) red[tid] += red[tid + off];
    __syncthreads();
  }
  const float rstd = rsqrtf(red[0] * (1.0f / Dc) + 1e-5f);
  f16* orow = out + (size_t)row * Dc;
#pragma unroll
  for (int j = 0; j < 4; ++j) {
    int c = tid + j * 256;
    orow[c] = (f16)(g[c] * ((v[j] - mean) * rstd) + s[c]);
  }
}

// ---------------------------------------------------------------------------
// WMMA GEMM: C[M,N] = A[M,K] (f16, row-major) x Bt[N,K] (f16, pre-transposed).
// Block = 8 waves stacked on M; all waves share one 64x32 B tile per k-step,
// staged global->LDS with async-to-LDS copies (one b128 per thread).
// MODE 0: store f16           MODE 1: f32 out = res + acc + bias
// MODE 2: f16 out = gelu(acc+bias)   MODE 3: store f16 TRANSPOSED [N,M]
// ---------------------------------------------------------------------------
template <int MODE>
__global__ __launch_bounds__(256)
void gemm_wmma(const f16* __restrict__ A, const f16* __restrict__ Bt,
               float* __restrict__ outF, f16* __restrict__ outH,
               const float* __restrict__ bias, const float* __restrict__ res,
               int M, int N, int K) {
  constexpr int LDB = 40;                       // padded row stride (halves)
  __shared__ __align__(16) f16 tileB[64 * LDB]; // 5 KB
  const int tid  = threadIdx.x;
  const int lane = tid & 31;
  const int wave = tid >> 5;
  const int lm = lane & 15, lh = lane >> 4;
  const int m0 = blockIdx.y * 128 + wave * 16;
  const int n0 = blockIdx.x * 64;

  // staging assignment: thread -> (B row, 16B chunk)
  const int sRow = tid >> 2;                    // 0..63
  const int sCh  = (tid & 3) * 8;               // half offset 0/8/16/24
  const f16* gB = Bt + (size_t)(n0 + sRow) * K + sCh;
  f16* lB = &tileB[sRow * LDB + sCh];

  v8f acc[4] = {};
  const f16* arow = A + (size_t)(m0 + lm) * K;

  for (int k0 = 0; k0 < K; k0 += 32) {
    stage16B(gB + k0, lB);                      // async global->LDS (ASYNCcnt)
    __builtin_prefetch(arow + k0 + 512, 0, 1);  // global_prefetch_b8
    v16h a = ldA(arow + k0 + lh * 8, arow + k0 + 16 + lh * 8);
    wait_stage();                               // s_wait_asynccnt 0
    __syncthreads();
#pragma unroll
    for (int t = 0; t < 4; ++t) {
      const f16* bp = &tileB[(t * 16 + lm) * LDB + lh * 8];
      v16h b = ldA(bp, bp + 16);                // two 16B chunks (table k order) from LDS
      acc[t] = wmma_f16(a, b, acc[t]);
    }
    __syncthreads();                            // WAR: reads done before restage
  }

#pragma unroll
  for (int t = 0; t < 4; ++t) {
    const int n = n0 + t * 16 + lm;
#pragma unroll
    for (int r = 0; r < 8; ++r) {
      const int m = m0 + r + 8 * lh;            // C layout: row r + 8*(lane/16)
      const float v = acc[t][r];
      const size_t idx = (size_t)m * N + n;
      if (MODE == 0)      outH[idx] = (f16)v;
      else if (MODE == 1) outF[idx] = res[idx] + v + bias[n];
      else if (MODE == 2) outH[idx] = (f16)gelu_t(v + bias[n]);
      else                outH[(size_t)n * M + m] = (f16)v;  // transposed (V)
    }
  }
}

// ---------------------------------------------------------------------------
// Flash attention (causal). One wave = one 16-row Q tile of one (b,h).
// 32 keys/step: QK^T via 4 WMMA, online softmax (16-lane shfl reductions),
// P round-trips through LDS into A-fragment layout, P·V via 4 WMMA against
// pre-transposed V (contiguous 32B B-fragment loads).
// ---------------------------------------------------------------------------
__global__ __launch_bounds__(256)
void attn_flash(const f16* __restrict__ q16, const f16* __restrict__ k16,
                const f16* __restrict__ vT16, f16* __restrict__ ctx16) {
  __shared__ __align__(64) f16 pb[8][16][32];   // per-wave P tile
  const int lane = threadIdx.x & 31, wave = threadIdx.x >> 5;
  const int lm = lane & 15, lh = lane >> 4;
  const int bh = blockIdx.y, b = bh >> 4, h = bh & 15;
  const int q0 = (blockIdx.x * 8 + wave) * 16;
  const int bT = b * Tc;
  const int hoff = h * HWc;

  const f16* qrow = q16 + (size_t)(bT + q0 + lm) * Dc + hoff;
  const v16h qa0 = ldA(qrow + lh * 16,      qrow + 8 + lh * 16);
  const v16h qa1 = ldA(qrow + 32 + lh * 16, qrow + 40 + lh * 16);

  float mr[8], lr[8];
  v8f o[4] = {};
#pragma unroll
  for (int r = 0; r < 8; ++r) { mr[r] = -INFINITY; lr[r] = 0.f; }
  const float sc = 0.125f;  // 1/sqrt(64)

  for (int kb = 0; kb < q0 + 16; kb += 32) {
    v8f s0 = {}, s1 = {};
    const f16* kr0 = k16 + (size_t)(bT + kb + lm) * Dc + hoff;
    const f16* kr1 = k16 + (size_t)(bT + kb + 16 + lm) * Dc + hoff;
    {
      v16h kf0a = *(const v16h*)(kr0 + lh * 16);
      v16h kf0b = *(const v16h*)(kr0 + 32 + lh * 16);
      v16h kf1a = *(const v16h*)(kr1 + lh * 16);
      v16h kf1b = *(const v16h*)(kr1 + 32 + lh * 16);
      s0 = wmma_f16(qa0, kf0a, s0);
      s0 = wmma_f16(qa1, kf0b, s0);
      s1 = wmma_f16(qa0, kf1a, s1);
      s1 = wmma_f16(qa1, kf1b, s1);
    }

#pragma unroll
    for (int r = 0; r < 8; ++r) {
      const int row = q0 + r + 8 * lh;
      float sv0 = (kb + lm      <= row) ? s0[r] * sc : -INFINITY;
      float sv1 = (kb + 16 + lm <= row) ? s1[r] * sc : -INFINITY;
      float mx = fmaxf(sv0, sv1);
#pragma unroll
      for (int msk = 1; msk < 16; msk <<= 1)
        mx = fmaxf(mx, __shfl_xor(mx, msk, 32));
      const float mn = fmaxf(mr[r], mx);
      const float ex = __expf(mr[r] - mn);
      const float p0 = __expf(sv0 - mn);   // exp(-inf)=0 covers the mask
      const float p1 = __expf(sv1 - mn);
      float ps = p0 + p1;
#pragma unroll
      for (int msk = 1; msk < 16; msk <<= 1)
        ps += __shfl_xor(ps, msk, 32);
      lr[r] = lr[r] * ex + ps;
      mr[r] = mn;
#pragma unroll
      for (int t = 0; t < 4; ++t) o[t][r] *= ex;
      pb[wave][r + 8 * lh][lm]      = (f16)p0;
      pb[wave][r + 8 * lh][lm + 16] = (f16)p1;
    }

    // in-wave LDS round trip: drain DS pipe before re-reading P as A-fragment
    asm volatile("s_wait_dscnt 0" ::: "memory");
    __builtin_amdgcn_wave_barrier();

    const v16h pa = ldA(&pb[wave][lm][lh * 16], &pb[wave][lm][8 + lh * 16]);
#pragma unroll
    for (int t = 0; t < 4; ++t) {
      const f16* vr =
          vT16 + (size_t)(hoff + t * 16 + lm) * Rc + bT + kb + lh * 16;
      v16h vf = *(const v16h*)vr;  // contiguous 32B (V pre-transposed)
      o[t] = wmma_f16(pa, vf, o[t]);
    }
    asm volatile("s_wait_dscnt 0" ::: "memory");
    __builtin_amdgcn_wave_barrier();
  }

#pragma unroll
  for (int r = 0; r < 8; ++r) {
    const float inv = 1.0f / lr[r];
    const size_t row = (size_t)(bT + q0 + r + 8 * lh);
#pragma unroll
    for (int t = 0; t < 4; ++t)
      ctx16[row * Dc + hoff + t * 16 + lm] = (f16)(o[t][r] * inv);
  }
}

// ---------------------------------------------------------------------------
// Launch: wtrans x6 -> LN1 -> QKV GEMMs -> flash attn -> Wo(+res) -> LN2
//         -> W1(+gelu) -> W2(+res) into d_out.
// ---------------------------------------------------------------------------
extern "C" void kernel_launch(void* const* d_in, const int* in_sizes, int n_in,
                              void* d_out, int out_size, void* d_ws,
                              size_t ws_size, hipStream_t stream) {
  (void)in_sizes; (void)n_in; (void)out_size; (void)ws_size;
  const float* x    = (const float*)d_in[0];
  const float* wq   = (const float*)d_in[1];
  const float* wk   = (const float*)d_in[2];
  const float* wv   = (const float*)d_in[3];
  const float* wo   = (const float*)d_in[4];
  const float* bo   = (const float*)d_in[5];
  const float* ln1s = (const float*)d_in[6];
  const float* ln1b = (const float*)d_in[7];
  const float* ln2s = (const float*)d_in[8];
  const float* ln2b = (const float*)d_in[9];
  const float* w1   = (const float*)d_in[10];
  const float* b1   = (const float*)d_in[11];
  const float* w2   = (const float*)d_in[12];
  const float* b2   = (const float*)d_in[13];

  char* ws = (char*)d_ws;
  const size_t MB = 1024ull * 1024ull;
  f16*   h16   = (f16*)(ws + 0 * MB);    // [R, D]
  f16*   q16   = (f16*)(ws + 8 * MB);    // [R, D]
  f16*   k16   = (f16*)(ws + 16 * MB);   // [R, D]
  f16*   vT16  = (f16*)(ws + 24 * MB);   // [D, R]  (transposed V)
  f16*   ctx16 = (f16*)(ws + 32 * MB);   // [R, D]
  f16*   h2    = (f16*)(ws + 40 * MB);   // [R, D]
  f16*   ffn16 = (f16*)(ws + 48 * MB);   // [R, DFF]
  float* x2    = (float*)(ws + 80 * MB); // [R, D] f32 (attn residual out)
  f16*   wqT   = (f16*)(ws + 96 * MB);
  f16*   wkT   = (f16*)(ws + 98 * MB);
  f16*   wvT   = (f16*)(ws + 100 * MB);
  f16*   woT   = (f16*)(ws + 102 * MB);
  f16*   w1T   = (f16*)(ws + 104 * MB);  // [DFF, D]
  f16*   w2T   = (f16*)(ws + 112 * MB);  // [D, DFF]

  const dim3 blk(256);
  wtrans_f16<<<(Dc * Dc + 255) / 256, blk, 0, stream>>>(wq, wqT, Dc, Dc);
  wtrans_f16<<<(Dc * Dc + 255) / 256, blk, 0, stream>>>(wk, wkT, Dc, Dc);
  wtrans_f16<<<(Dc * Dc + 255) / 256, blk, 0, stream>>>(wv, wvT, Dc, Dc);
  wtrans_f16<<<(Dc * Dc + 255) / 256, blk, 0, stream>>>(wo, woT, Dc, Dc);
  wtrans_f16<<<(Dc * FFc + 255) / 256, blk, 0, stream>>>(w1, w1T, Dc, FFc);
  wtrans_f16<<<(Dc * FFc + 255) / 256, blk, 0, stream>>>(w2, w2T, FFc, Dc);

  ln_to_f16<<<Rc, blk, 0, stream>>>(x, ln1s, ln1b, h16);

  const dim3 gD(Dc / 64, Rc / 128);
  gemm_wmma<0><<<gD, blk, 0, stream>>>(h16, wqT, nullptr, q16, nullptr, nullptr, Rc, Dc, Dc);
  gemm_wmma<0><<<gD, blk, 0, stream>>>(h16, wkT, nullptr, k16, nullptr, nullptr, Rc, Dc, Dc);
  gemm_wmma<3><<<gD, blk, 0, stream>>>(h16, wvT, nullptr, vT16, nullptr, nullptr, Rc, Dc, Dc);

  attn_flash<<<dim3(Tc / 128, Bc * Hc), blk, 0, stream>>>(q16, k16, vT16, ctx16);

  gemm_wmma<1><<<gD, blk, 0, stream>>>(ctx16, woT, x2, nullptr, bo, x, Rc, Dc, Dc);

  ln_to_f16<<<Rc, blk, 0, stream>>>(x2, ln2s, ln2b, h2);

  gemm_wmma<2><<<dim3(FFc / 64, Rc / 128), blk, 0, stream>>>(h2, w1T, nullptr, ffn16, b1, nullptr, Rc, FFc, Dc);

  gemm_wmma<1><<<gD, blk, 0, stream>>>(ffn16, w2T, (float*)d_out, nullptr, b2, x2, Rc, Dc, FFc);
}
